// Attention_65816078844372
// MI455X (gfx1250) — compile-verified, matrix operand order checked by reading, not run on hardware
//
#include <hip/hip_runtime.h>

typedef __attribute__((ext_vector_type(16))) __bf16 v16bf;
typedef __attribute__((ext_vector_type(8)))  float  v8f;

#define WMMA_BF16(a, b, c) \
  __builtin_amdgcn_wmma_f32_16x16x32_bf16(false, (a), false, (b), (short)0, (c), false, false)

static constexpr int BATCH = 4;
static constexpr int SEQ   = 2048;
static constexpr int FEAT  = 1024;
static constexpr int HEADS = 16;
static constexpr int HDIM  = 64;
static constexpr int MROWS = BATCH * SEQ;   // 8192

// ---------------------------------------------------------------------------
// f32 -> bf16 conversion (vectorized x4)
// ---------------------------------------------------------------------------
__global__ __launch_bounds__(256) void cvt_f32_bf16(const float* __restrict__ in,
                                                    __bf16* __restrict__ out, int n) {
  int i = (blockIdx.x * 256 + threadIdx.x) * 4;
  if (i + 3 < n) {
    float4 f = *(const float4*)(in + i);
    out[i + 0] = (__bf16)f.x;
    out[i + 1] = (__bf16)f.y;
    out[i + 2] = (__bf16)f.z;
    out[i + 3] = (__bf16)f.w;
  }
}

// ---------------------------------------------------------------------------
// QKV projection GEMM: [8192 x 1024] @ [1024 x 1024] + bias, bf16 in, bf16 out
// Output written in head-split layout [B, H, N, D].
// Block: 256 threads (8 waves). Tile: BM=128, BN=64, BK=64.
// Wave (w): wm = w&3 (4 x 32-row strips), wn = w>>2 (2 x 32-col strips).
// 8 WMMAs per staged tile per wave (2x2 tiles x 2 k-steps).
// ---------------------------------------------------------------------------
__global__ __launch_bounds__(256) void qkv_gemm(
    const __bf16* __restrict__ Xb,
    const __bf16* __restrict__ Wqb, const __bf16* __restrict__ Wkb, const __bf16* __restrict__ Wvb,
    const float* __restrict__ bq, const float* __restrict__ bk, const float* __restrict__ bv,
    __bf16* __restrict__ Qo, __bf16* __restrict__ Ko, __bf16* __restrict__ Vo) {

  const int zz = blockIdx.z;
  const __bf16* W   = (zz == 0) ? Wqb : (zz == 1) ? Wkb : Wvb;
  const float* bias = (zz == 0) ? bq  : (zz == 1) ? bk  : bv;
  __bf16* Out       = (zz == 0) ? Qo  : (zz == 1) ? Ko  : Vo;

  const int m0 = blockIdx.x * 128;
  const int n0 = blockIdx.y * 64;

  __shared__ __align__(32) __bf16 ldsA[128 * 64];  // [m][k] row-major, 16KB
  __shared__ __align__(32) __bf16 ldsB[64 * 64];   // [n][k] (W transposed), 8KB

  const int t     = threadIdx.x;
  const int w     = t >> 5;
  const int lane  = t & 31;
  const int lrow  = lane & 15;
  const int lhalf = lane >> 4;
  const int wm    = w & 3;
  const int wn    = w >> 2;

  v8f acc[2][2] = {};

  for (int k0 = 0; k0 < FEAT; k0 += 64) {
    // --- stage A: 128x64 bf16, 32 elements (64B) per thread, coalesced ---
    {
      const int row  = t >> 1;
      const int half = t & 1;
      const __bf16* src = Xb + (size_t)(m0 + row) * FEAT + k0 + half * 32;
      __bf16* dst = ldsA + row * 64 + half * 32;
      ((uint4*)dst)[0] = ((const uint4*)src)[0];
      ((uint4*)dst)[1] = ((const uint4*)src)[1];
      ((uint4*)dst)[2] = ((const uint4*)src)[2];
      ((uint4*)dst)[3] = ((const uint4*)src)[3];
      if (k0 + 64 < FEAT)
        __builtin_prefetch(src + 64, 0, 1);  // global_prefetch_b8 next K tile
    }
    // --- stage B transposed: 2 rows of 8 consecutive bf16 -> [n][k] ---
    {
      const int kk = t >> 3;          // 0..31 (and +32)
      const int nc = (t & 7) * 8;     // 0..56
#pragma unroll
      for (int kh = 0; kh < 2; kh++) {
        const int k = kk + kh * 32;
        const __bf16* src = W + (size_t)(k0 + k) * FEAT + n0 + nc;
        __bf16 tmp[8];
        *(uint4*)tmp = *(const uint4*)src;   // 8 bf16 = 16 bytes, coalesced
#pragma unroll
        for (int i = 0; i < 8; i++)
          ldsB[(nc + i) * 64 + k] = tmp[i];
      }
    }
    __syncthreads();

    // --- compute: 2 k-steps of 32, 2x2 WMMA tiles each ---
#pragma unroll
    for (int ks = 0; ks < 2; ks++) {
      v16bf af[2], bf[2];
#pragma unroll
      for (int i = 0; i < 2; i++)
        af[i] = *(const v16bf*)(ldsA + (wm * 32 + i * 16 + lrow) * 64 +
                                ks * 32 + lhalf * 16);
#pragma unroll
      for (int j = 0; j < 2; j++)
        bf[j] = *(const v16bf*)(ldsB + (wn * 32 + j * 16 + lrow) * 64 +
                                ks * 32 + lhalf * 16);
#pragma unroll
      for (int i = 0; i < 2; i++)
#pragma unroll
        for (int j = 0; j < 2; j++)
          acc[i][j] = WMMA_BF16(af[i], bf[j], acc[i][j]);
    }
    __syncthreads();
  }

  // --- epilogue: + bias, write bf16 head-split [B,H,N,D] ---
#pragma unroll
  for (int i = 0; i < 2; i++) {
#pragma unroll
    for (int j = 0; j < 2; j++) {
      const int c = n0 + wn * 32 + j * 16 + lrow;   // feature column
      const int h = c >> 6, d = c & 63;
      const float bv_ = bias[c];
#pragma unroll
      for (int r = 0; r < 8; r++) {
        const int m = m0 + wm * 32 + i * 16 + r + lhalf * 8;  // global row
        const int b = m >> 11, n = m & 2047;
        const float v = acc[i][j][r] + bv_;
        Out[(((size_t)(b * HEADS + h)) * SEQ + n) * HDIM + d] = (__bf16)v;
      }
    }
  }
}

// ---------------------------------------------------------------------------
// Flash attention with causal mask.
// Grid: (SEQ/64, B*H). Block: 128 threads = 4 waves, each wave owns a 16-row
// Q tile. Key blocks of 64: 8 S-WMMAs + 8 PV-WMMAs per 3 barriers.
// Online softmax in C-matrix register layout (row M = r + 8*(lane>=16)).
// ---------------------------------------------------------------------------
__global__ __launch_bounds__(128) void attn_fwd(
    const __bf16* __restrict__ Q, const __bf16* __restrict__ K,
    const __bf16* __restrict__ V, float* __restrict__ out) {

  const int bh = blockIdx.y;               // b*16 + h
  const int qb = blockIdx.x * 64;          // block's first query row
  const size_t head_off = (size_t)bh * SEQ * HDIM;
  const __bf16* Qp = Q + head_off;
  const __bf16* Kp = K + head_off;
  const __bf16* Vp = V + head_off;

  const int t     = threadIdx.x;
  const int w     = t >> 5;
  const int lane  = t & 31;
  const int lrow  = lane & 15;
  const int lhalf = lane >> 4;
  const int qrow0 = qb + w * 16;           // wave's q rows: qrow0 .. qrow0+15

  __shared__ __align__(32) __bf16 ldsVT[64 * 64];     // V^T: [d][key], 8KB
  __shared__ __align__(32) __bf16 ldsP[4 * 16 * 64];  // per-wave P tile [m][64], 8KB
  __bf16* Pw = ldsP + w * 1024;

  // Q A-frags (16x64 = 2 chunks of 16x32), contiguous global loads
  v16bf qf[2];
#pragma unroll
  for (int kd = 0; kd < 2; kd++)
    qf[kd] = *(const v16bf*)(Qp + (size_t)(qrow0 + lrow) * HDIM + kd * 32 + lhalf * 16);

  v8f o[4] = {};                            // O: 4 d-tiles of 16x16 f32
  float m_r[8], l_r[8];
#pragma unroll
  for (int r = 0; r < 8; r++) { m_r[r] = -1e30f; l_r[r] = 0.0f; }

  const int jend = qb + 64;                 // causal: keys 0 .. qb+63
  for (int jc = 0; jc < jend; jc += 64) {
    // --- stage V^T tile (64 keys x 64 dims -> [d][key]) cooperatively ---
    {
      const int koff = t >> 1;              // 0..63 key within block
      const int dch  = (t & 1) * 32;        // 0 / 32
      const __bf16* vs = Vp + (size_t)(jc + koff) * HDIM + dch;
      __bf16 tmp[32];
      ((uint4*)tmp)[0] = ((const uint4*)vs)[0];
      ((uint4*)tmp)[1] = ((const uint4*)vs)[1];
      ((uint4*)tmp)[2] = ((const uint4*)vs)[2];
      ((uint4*)tmp)[3] = ((const uint4*)vs)[3];
#pragma unroll
      for (int i = 0; i < 32; i++)
        ldsVT[(dch + i) * 64 + koff] = tmp[i];
    }
    __syncthreads();

    // --- S = Q . K^T for four 16-key tiles; K^T B-frags are contiguous ---
    v8f s[4] = {};
#pragma unroll
    for (int jt = 0; jt < 4; jt++) {
#pragma unroll
      for (int kd = 0; kd < 2; kd++) {
        v16bf kf = *(const v16bf*)(Kp + (size_t)(jc + jt * 16 + lrow) * HDIM +
                                   kd * 32 + lhalf * 16);
        s[jt] = WMMA_BF16(qf[kd], kf, s[jt]);
      }
    }

    // --- scale, causal mask, online softmax (per C-layout row groups) ---
#pragma unroll
    for (int r = 0; r < 8; r++) {
      const int qg = qrow0 + r + lhalf * 8;
      float sv[4];
#pragma unroll
      for (int jt = 0; jt < 4; jt++) {
        float v = s[jt][r] * 0.125f;
        if (jc + jt * 16 + lrow > qg) v = -1e30f;
        sv[jt] = v;
      }

      float rowm = fmaxf(fmaxf(sv[0], sv[1]), fmaxf(sv[2], sv[3]));
#pragma unroll
      for (int off = 1; off < 16; off <<= 1)
        rowm = fmaxf(rowm, __shfl_xor(rowm, off, 32));

      const float mnew = fmaxf(m_r[r], rowm);
      const float corr = __expf(m_r[r] - mnew);
      m_r[r] = mnew;

      float p[4], rsum = 0.0f;
#pragma unroll
      for (int jt = 0; jt < 4; jt++) {
        p[jt] = __expf(sv[jt] - mnew);
        rsum += p[jt];
      }
#pragma unroll
      for (int off = 1; off < 16; off <<= 1)
        rsum += __shfl_xor(rsum, off, 32);
      l_r[r] = l_r[r] * corr + rsum;

#pragma unroll
      for (int nt = 0; nt < 4; nt++) o[nt][r] *= corr;

      // store P in A-frag friendly [m][64] layout
#pragma unroll
      for (int jt = 0; jt < 4; jt++)
        Pw[(r + lhalf * 8) * 64 + jt * 16 + lrow] = (__bf16)p[jt];
    }
    __syncthreads();

    // --- O += P(16x64) . V(64x64): 2 k-steps x 4 d-tiles ---
#pragma unroll
    for (int kp = 0; kp < 2; kp++) {
      v16bf pf = *(const v16bf*)(Pw + lrow * 64 + kp * 32 + lhalf * 16);
#pragma unroll
      for (int nt = 0; nt < 4; nt++) {
        v16bf vf = *(const v16bf*)(ldsVT + (nt * 16 + lrow) * 64 +
                                   kp * 32 + lhalf * 16);
        o[nt] = WMMA_BF16(pf, vf, o[nt]);
      }
    }
    __syncthreads();
  }

  // --- epilogue: normalize and write f32 [B, N, F] (merged heads) ---
  const int b = bh >> 4, h = bh & 15;
#pragma unroll
  for (int nt = 0; nt < 4; nt++) {
    const int d = nt * 16 + lrow;
#pragma unroll
    for (int r = 0; r < 8; r++) {
      const int qg = qrow0 + r + lhalf * 8;
      const float val = o[nt][r] / l_r[r];
      out[((size_t)b * SEQ + qg) * FEAT + h * HDIM + d] = val;
    }
  }
}

// ---------------------------------------------------------------------------
extern "C" void kernel_launch(void* const* d_in, const int* in_sizes, int n_in,
                              void* d_out, int out_size, void* d_ws, size_t ws_size,
                              hipStream_t stream) {
  (void)in_sizes; (void)n_in; (void)out_size; (void)ws_size;
  const float* x  = (const float*)d_in[0];
  const float* Wq = (const float*)d_in[1];
  const float* bq = (const float*)d_in[2];
  const float* Wk = (const float*)d_in[3];
  const float* bk = (const float*)d_in[4];
  const float* Wv = (const float*)d_in[5];
  const float* bv = (const float*)d_in[6];
  float* out = (float*)d_out;

  char* ws = (char*)d_ws;
  // workspace layout (bytes):
  //   Xb   @ 0         : 8192*1024*2   = 16,777,216
  //   Wqb  @ 16777216  : 1024*1024*2   =  2,097,152
  //   Wkb  @ +2MB, Wvb @ +4MB
  //   Qb   @ 23068672  : 8M elems bf16 = 16,777,216   (layout [B,H,N,D])
  //   Kb, Vb follow    : total 73,400,320 bytes
  __bf16* Xb  = (__bf16*)(ws);
  __bf16* Wqb = (__bf16*)(ws + 16777216);
  __bf16* Wkb = (__bf16*)(ws + 16777216 + 2097152);
  __bf16* Wvb = (__bf16*)(ws + 16777216 + 2 * 2097152);
  __bf16* Qb  = (__bf16*)(ws + 23068672);
  __bf16* Kb  = (__bf16*)(ws + 23068672 + 16777216);
  __bf16* Vb  = (__bf16*)(ws + 23068672 + 2 * 16777216);

  cvt_f32_bf16<<<8192, 256, 0, stream>>>(x,  Xb,  MROWS * FEAT);
  cvt_f32_bf16<<<1024, 256, 0, stream>>>(Wq, Wqb, FEAT * FEAT);
  cvt_f32_bf16<<<1024, 256, 0, stream>>>(Wk, Wkb, FEAT * FEAT);
  cvt_f32_bf16<<<1024, 256, 0, stream>>>(Wv, Wvb, FEAT * FEAT);

  qkv_gemm<<<dim3(MROWS / 128, FEAT / 64, 3), 256, 0, stream>>>(
      Xb, Wqb, Wkb, Wvb, bq, bk, bv, Qb, Kb, Vb);

  attn_fwd<<<dim3(SEQ / 64, BATCH * HEADS), 128, 0, stream>>>(Qb, Kb, Vb, out);
}
